// EncoderLayer_23502061043745
// MI455X (gfx1250) — compile-verified, matrix operand order checked by reading, not run on hardware
//
#include <hip/hip_runtime.h>
#include <hip/hip_bf16.h>

// ---------------------------------------------------------------------------
// CDNA5 / gfx1250 transformer encoder layer.
// bf16 WMMA for all GEMMs; TDM (tensor_load_to_lds) double-buffered staging.
// B=4, S=2048, D=1024, H=16, DH=64, DFF=4096.
// ---------------------------------------------------------------------------

typedef __attribute__((ext_vector_type(16))) __bf16       v16bf;
typedef __attribute__((ext_vector_type(8)))  __bf16       v8bf;
typedef __attribute__((ext_vector_type(8)))  float        v8f;
typedef __attribute__((ext_vector_type(4)))  unsigned int v4u;
typedef __attribute__((ext_vector_type(8)))  int          v8i;
typedef __attribute__((ext_vector_type(4)))  int          v4i;

static __device__ __forceinline__ v8f wmma_bf16(v16bf a, v16bf b, v8f c) {
  return __builtin_amdgcn_wmma_f32_16x16x32_bf16(false, a, false, b, (short)0, c,
                                                 false, false);
}

// 16x32 A-type bf16 fragment from LDS (CDNA5 ISA 7.12.2 layout):
// lane L: row M = L%16; K = 8*(L/16) + h%8 + 16*(h/8)  -> two 16B runs.
// B operands use the same helper on an LDS tile stored [n][k].
static __device__ __forceinline__ v16bf load_frag(const __bf16* base, int row0,
                                                  int stride_h, int k0, int lane) {
  const int m   = lane & 15;
  const int klo = k0 + ((lane >> 4) << 3);
  const __bf16* p = base + (size_t)(row0 + m) * stride_h + klo;
  v8bf lo = *(const v8bf*)(p);
  v8bf hi = *(const v8bf*)(p + 16);
  v16bf f;
#pragma unroll
  for (int i = 0; i < 8; ++i) { f[i] = lo[i]; f[8 + i] = hi[i]; }
  return f;
}

// ---------------------------------------------------------------------------
// Tensor Data Mover: 2D bf16 tile load global -> LDS with hardware row padding.
// D# packing per CDNA5 ISA (group0 128b, group1 256b).
// pad_interval code c => pad after 2^(c+1) DWORDs; pad_amount code a => (a+1) DWORDs.
// ---------------------------------------------------------------------------
typedef __attribute__((address_space(3))) const void* lds_cptr_t;
static __device__ __forceinline__ unsigned lds_off(const void* p) {
  return (unsigned)(unsigned long long)(lds_cptr_t)p;
}

static __device__ __forceinline__ void tdm_load_2d(
    unsigned lds_addr, const void* gptr,
    unsigned tile_d0 /*elems*/, unsigned tile_d1 /*rows*/,
    unsigned long long row_stride /*elems*/,
    unsigned pad_interval_code, unsigned pad_amount_code) {
  const unsigned long long ga = (unsigned long long)gptr;
  v4u g0;
  g0[0] = 1u;                                             // count=1, user D#
  g0[1] = lds_addr;                                       // lds_addr [63:32]
  g0[2] = (unsigned)ga;                                   // global_addr lo
  g0[3] = (unsigned)((ga >> 32) & 0x01FFFFFFull) | (2u << 30);  // addr hi | type=2
  const unsigned flags = (1u << 16)                        // data_size = 2 bytes
                       | (1u << 20)                        // pad_enable
                       | (pad_interval_code << 22)
                       | (pad_amount_code << 25);
  v8i g1;
  g1[0] = (int)flags;
  g1[1] = (int)((tile_d0 & 0xFFFFu) << 16);                       // tensor_dim0 lo16
  g1[2] = (int)((tile_d0 >> 16) | ((tile_d1 & 0xFFFFu) << 16));   // dim0 hi | dim1 lo
  g1[3] = (int)((tile_d1 >> 16) | (tile_d0 << 16));               // dim1 hi | tile_dim0
  g1[4] = (int)(tile_d1 & 0xFFFFu);                               // tile_dim1 (dim2=0)
  g1[5] = (int)(unsigned)(row_stride & 0xFFFFFFFFull);            // dim0_stride lo32
  g1[6] = (int)(unsigned)((row_stride >> 32) & 0xFFFFull);        // dim0_stride hi16
  g1[7] = 0;
  v4i gz = {0, 0, 0, 0};
#if defined(__clang_major__) && (__clang_major__ >= 23)
  v8i gz8 = {0, 0, 0, 0, 0, 0, 0, 0};
  __builtin_amdgcn_tensor_load_to_lds(g0, g1, gz, gz, gz8, 0);
#else
  __builtin_amdgcn_tensor_load_to_lds(g0, g1, gz, gz, 0);
#endif
}

// ---------------------------------------------------------------------------
// fp32 -> bf16 conversion (grid-stride)
// ---------------------------------------------------------------------------
__global__ void cvt_f32_bf16(const float* __restrict__ in, __bf16* __restrict__ out,
                             size_t n) {
  size_t i = (size_t)blockIdx.x * blockDim.x + threadIdx.x;
  size_t stride = (size_t)gridDim.x * blockDim.x;
  for (; i < n; i += stride) out[i] = (__bf16)in[i];
}

// fp32 -> bf16 transpose: in z-batch is R x C, out is C x R. R,C multiples of 32.
__global__ __launch_bounds__(256) void cvt_transpose(const float* __restrict__ in,
                                                     __bf16* __restrict__ out,
                                                     int R, int C) {
  __shared__ float t[32][33];
  const size_t zoff = (size_t)blockIdx.z * R * C;
  const int r0 = blockIdx.y * 32, c0 = blockIdx.x * 32;
  const int tx = threadIdx.x & 31, ty = threadIdx.x >> 5;   // 32 x 8
#pragma unroll
  for (int i = 0; i < 4; ++i)
    t[ty + i * 8][tx] = in[zoff + (size_t)(r0 + ty + i * 8) * C + c0 + tx];
  __syncthreads();
#pragma unroll
  for (int i = 0; i < 4; ++i)
    out[zoff + (size_t)(c0 + ty + i * 8) * R + r0 + tx] = (__bf16)t[tx][ty + i * 8];
}

// ---------------------------------------------------------------------------
// QKV GEMM (64x64 tile, plain staging): Out = A[m][k] * Bt[n][k]^T + bias[n].
// Batched over z (z1 = batch, z2 = head). OUT_TRANS stores Out[n][m] (for V^T).
// ---------------------------------------------------------------------------
template <bool OUT_TRANS>
__global__ __launch_bounds__(128) void gemm_qkv(
    const __bf16* __restrict__ A, const __bf16* __restrict__ Bt,
    const float* __restrict__ bias, __bf16* __restrict__ Out,
    int K, int lda, int ldbt, int ldo, int z2n,
    long as1, long as2, long bs2, long bias2, long os1, long os2) {
  __shared__ __bf16 Asm[64 * 40];
  __shared__ __bf16 Bsm[64 * 40];

  const int tid = threadIdx.x, lane = tid & 31, w = tid >> 5;
  const int row0 = blockIdx.x * 64;
  const int z = blockIdx.z, z1 = z / z2n, z2 = z - z1 * z2n;

  const __bf16* Ap = A + (size_t)z1 * as1 + (size_t)z2 * as2 + (size_t)row0 * lda;
  const __bf16* Bp = Bt + (size_t)z2 * bs2;
  const float*  bp = bias + (size_t)z2 * bias2;
  const size_t  obase = (size_t)z1 * os1 + (size_t)z2 * os2;

  v8f acc[4];
#pragma unroll
  for (int t = 0; t < 4; ++t)
#pragma unroll
    for (int r = 0; r < 8; ++r) acc[t][r] = 0.f;

  for (int k0 = 0; k0 < K; k0 += 32) {
#pragma unroll
    for (int i = 0; i < 2; ++i) {
      int c = tid + 128 * i, r = c >> 2, kk = (c & 3) * 8;
      *(v8bf*)&Asm[r * 40 + kk] = *(const v8bf*)&Ap[(size_t)r * lda + k0 + kk];
      *(v8bf*)&Bsm[r * 40 + kk] = *(const v8bf*)&Bp[(size_t)r * ldbt + k0 + kk];
    }
    __syncthreads();
    v16bf bf = load_frag(Bsm, w * 16, 40, 0, lane);
#pragma unroll
    for (int t = 0; t < 4; ++t) {
      v16bf af = load_frag(Asm, t * 16, 40, 0, lane);
      acc[t] = wmma_bf16(af, bf, acc[t]);
    }
    __syncthreads();
  }

  const int   nl = w * 16 + (lane & 15);
  const float bv = bp[nl];
#pragma unroll
  for (int t = 0; t < 4; ++t) {
#pragma unroll
    for (int r = 0; r < 8; ++r) {
      int   m = row0 + t * 16 + r + ((lane >> 4) << 3);
      float v = acc[t][r] + bv;
      if (OUT_TRANS)
        Out[obase + (size_t)nl * ldo + m] = (__bf16)v;
      else
        Out[obase + (size_t)m * ldo + nl] = (__bf16)v;
    }
  }
}

// ---------------------------------------------------------------------------
// Big GEMM: 128x128 tile, BK=32, 256 threads = 8 waves (2x4), TDM double-buffered.
// Out[m][n] = A[m][k] * Bt[n][k]^T + bias[n]; optional ReLU; f32 or bf16 out.
// ---------------------------------------------------------------------------
template <bool RELU, bool OUT_BF16>
__global__ __launch_bounds__(256) void gemm128(
    const __bf16* __restrict__ A, const __bf16* __restrict__ Bt,
    const float* __restrict__ bias, void* __restrict__ Out,
    int K, int lda, int ldbt, int ldo) {
  __shared__ __bf16 Asm[2][128 * 40];
  __shared__ __bf16 Bsm[2][128 * 40];

  const int tid = threadIdx.x, lane = tid & 31, w = tid >> 5;
  const int wm = w >> 2, wn = w & 3;
  const int row0 = blockIdx.x * 128, col0 = blockIdx.y * 128;

  const __bf16* Ap = A + (size_t)row0 * lda;
  const __bf16* Bp = Bt + (size_t)col0 * ldbt;

  const unsigned aoff[2] = {lds_off(&Asm[0][0]), lds_off(&Asm[1][0])};
  const unsigned boff[2] = {lds_off(&Bsm[0][0]), lds_off(&Bsm[1][0])};
  const int nk = K / 32;

  // Prologue: DMA tile 0 (pad: after 32 halves (16 DW, code 3) insert 4 DW (code 3)
  // -> LDS row stride 40 halves).
  if (tid < 32) {
    tdm_load_2d(aoff[0], Ap, 32, 128, (unsigned long long)lda, 3, 3);
    tdm_load_2d(boff[0], Bp, 32, 128, (unsigned long long)ldbt, 3, 3);
  }

  v8f acc[4][2];
#pragma unroll
  for (int t = 0; t < 4; ++t)
#pragma unroll
    for (int j = 0; j < 2; ++j)
#pragma unroll
      for (int r = 0; r < 8; ++r) acc[t][j][r] = 0.f;

  for (int i = 0; i < nk; ++i) {
    __syncthreads();  // everyone done reading buf[(i+1)&1] (from iter i-1)
    if (tid < 32) {
      if (i + 1 < nk) {
        tdm_load_2d(aoff[(i + 1) & 1], Ap + (size_t)(i + 1) * 32, 32, 128,
                    (unsigned long long)lda, 3, 3);
        tdm_load_2d(boff[(i + 1) & 1], Bp + (size_t)(i + 1) * 32, 32, 128,
                    (unsigned long long)ldbt, 3, 3);
        __builtin_amdgcn_s_wait_tensorcnt((short)2);  // tile i complete (in-order)
      } else {
        __builtin_amdgcn_s_wait_tensorcnt((short)0);
      }
    }
    __syncthreads();  // publish tile i
    const __bf16* As = &Asm[i & 1][0];
    const __bf16* Bs = &Bsm[i & 1][0];
    v16bf bf0 = load_frag(Bs, wn * 32, 40, 0, lane);
    v16bf bf1 = load_frag(Bs, wn * 32 + 16, 40, 0, lane);
#pragma unroll
    for (int t = 0; t < 4; ++t) {
      v16bf af = load_frag(As, wm * 64 + t * 16, 40, 0, lane);
      acc[t][0] = wmma_bf16(af, bf0, acc[t][0]);
      acc[t][1] = wmma_bf16(af, bf1, acc[t][1]);
    }
  }

  const int   nb = col0 + wn * 32 + (lane & 15);
  const float bv0 = bias[nb], bv1 = bias[nb + 16];
#pragma unroll
  for (int t = 0; t < 4; ++t) {
#pragma unroll
    for (int r = 0; r < 8; ++r) {
      int   m = row0 + wm * 64 + t * 16 + r + ((lane >> 4) << 3);
      float v0 = acc[t][0][r] + bv0, v1 = acc[t][1][r] + bv1;
      if (RELU) { v0 = fmaxf(v0, 0.f); v1 = fmaxf(v1, 0.f); }
      if (OUT_BF16) {
        ((__bf16*)Out)[(size_t)m * ldo + nb]      = (__bf16)v0;
        ((__bf16*)Out)[(size_t)m * ldo + nb + 16] = (__bf16)v1;
      } else {
        ((float*)Out)[(size_t)m * ldo + nb]      = v0;
        ((float*)Out)[(size_t)m * ldo + nb + 16] = v1;
      }
    }
  }
}

// ---------------------------------------------------------------------------
// Flash attention, TDM double-buffered K / V^T tiles.
// One block = 64 query rows of one (b,h); 128 threads = 4 waves.
// Vt layout: (b,h,dh,s) so the V tile is a plain row tile (rows = dh).
// ---------------------------------------------------------------------------
__global__ __launch_bounds__(128) void attn_fwd(
    const __bf16* __restrict__ Qg, const __bf16* __restrict__ Kg,
    const __bf16* __restrict__ Vt, const int* __restrict__ maskg,
    __bf16* __restrict__ Og) {
  constexpr int S = 2048, DH = 64, D = 1024;
  __shared__ __bf16 Qs[64 * 72];        // [q][dh]
  __shared__ __bf16 Ks[2][64 * 72];     // [key][dh]
  __shared__ __bf16 Vs[2][64 * 72];     // [dh][key]
  __shared__ __bf16 Ps[4][16 * 72];     // per-wave P
  __shared__ int    mk[64];

  const int tid = threadIdx.x, lane = tid & 31, w = tid >> 5;
  const int qt = blockIdx.x, bh = blockIdx.y;
  const int b = bh >> 4, h = bh & 15;
  const size_t base = (size_t)bh * S * DH;

  const __bf16* Kbase  = Kg + base;                       // row stride DH
  const __bf16* Vtbase = Vt + (size_t)bh * DH * S;        // row stride S

  // Stage Q (plain loads, once).
#pragma unroll
  for (int i = 0; i < 4; ++i) {
    int c = tid + 128 * i, r = c >> 3, kk = (c & 7) * 8;
    *(v8bf*)&Qs[r * 72 + kk] =
        *(const v8bf*)&Qg[base + (size_t)(qt * 64 + r) * DH + kk];
  }
  const unsigned koff[2] = {lds_off(&Ks[0][0]), lds_off(&Ks[1][0])};
  const unsigned voff[2] = {lds_off(&Vs[0][0]), lds_off(&Vs[1][0])};
  // Prologue DMA: 64x64 tiles; pad after 64 halves (32 DW, code 4) by 4 DW (code 3)
  // -> LDS row stride 72 halves.
  if (tid < 32) {
    tdm_load_2d(koff[0], Kbase, 64, 64, DH, 4, 3);
    tdm_load_2d(voff[0], Vtbase, 64, 64, S, 4, 3);
  }
  __syncthreads();
  v16bf qa[2];
  qa[0] = load_frag(Qs, w * 16, 72, 0, lane);
  qa[1] = load_frag(Qs, w * 16, 72, 32, lane);

  float mrow[8], lrow[8];
  v8f   o[4];
#pragma unroll
  for (int r = 0; r < 8; ++r) { mrow[r] = -3e38f; lrow[r] = 0.f; }
#pragma unroll
  for (int t = 0; t < 4; ++t)
#pragma unroll
    for (int r = 0; r < 8; ++r) o[t][r] = 0.f;

  for (int kb = 0; kb < S / 64; ++kb) {
    __syncthreads();  // all waves done with buf[(kb+1)&1]
    if (tid < 32) {
      if (kb + 1 < S / 64) {
        tdm_load_2d(koff[(kb + 1) & 1], Kbase + (size_t)(kb + 1) * 64 * DH,
                    64, 64, DH, 4, 3);
        tdm_load_2d(voff[(kb + 1) & 1], Vtbase + (size_t)(kb + 1) * 64,
                    64, 64, S, 4, 3);
        __builtin_amdgcn_s_wait_tensorcnt((short)2);
      } else {
        __builtin_amdgcn_s_wait_tensorcnt((short)0);
      }
    }
    if (tid < 64) mk[tid] = maskg[(size_t)b * S + kb * 64 + tid];
    __syncthreads();  // publish tile kb + mask
    const __bf16* Kt = &Ks[kb & 1][0];
    const __bf16* Vtile = &Vs[kb & 1][0];

    // Scores: 16 q-rows x 64 keys.
    float sc[4][8];
#pragma unroll
    for (int nt = 0; nt < 4; ++nt) {
      v8f s;
#pragma unroll
      for (int r = 0; r < 8; ++r) s[r] = 0.f;
      s = wmma_bf16(qa[0], load_frag(Kt, nt * 16, 72, 0, lane), s);
      s = wmma_bf16(qa[1], load_frag(Kt, nt * 16, 72, 32, lane), s);
      const int masked = mk[nt * 16 + (lane & 15)];
#pragma unroll
      for (int r = 0; r < 8; ++r) sc[nt][r] = masked ? -1e9f : s[r] * 0.125f;
    }

    // Online softmax (row m = r + 8*(lane/16) lives in 16 lanes; width-16 xor reduce).
#pragma unroll
    for (int r = 0; r < 8; ++r) {
      float nm = mrow[r];
#pragma unroll
      for (int nt = 0; nt < 4; ++nt) nm = fmaxf(nm, sc[nt][r]);
#pragma unroll
      for (int off = 8; off > 0; off >>= 1)
        nm = fmaxf(nm, __shfl_xor(nm, off, 16));
      const float alpha = __expf(mrow[r] - nm);
      mrow[r] = nm;
      float ps = 0.f;
      const int prow = (r + ((lane >> 4) << 3)) * 72;
#pragma unroll
      for (int nt = 0; nt < 4; ++nt) {
        float p = __expf(sc[nt][r] - nm);
        ps += p;
        Ps[w][prow + nt * 16 + (lane & 15)] = (__bf16)p;
      }
#pragma unroll
      for (int off = 8; off > 0; off >>= 1) ps += __shfl_xor(ps, off, 16);
      lrow[r] = lrow[r] * alpha + ps;
#pragma unroll
      for (int t = 0; t < 4; ++t) o[t][r] *= alpha;
    }

    // O += P(16x64) x V(64x64); Ps is wave-private (same-wave LDS ops in order).
#pragma unroll
    for (int ks = 0; ks < 2; ++ks) {
      v16bf pa = load_frag(&Ps[w][0], 0, 72, ks * 32, lane);
#pragma unroll
      for (int dt = 0; dt < 4; ++dt)
        o[dt] = wmma_bf16(pa, load_frag(Vtile, dt * 16, 72, ks * 32, lane), o[dt]);
    }
  }

  // Normalize, scatter (b, s, h*64+e) bf16.
#pragma unroll
  for (int r = 0; r < 8; ++r) {
    const float inv  = 1.0f / lrow[r];
    const int   srow = qt * 64 + w * 16 + r + ((lane >> 4) << 3);
#pragma unroll
    for (int dt = 0; dt < 4; ++dt) {
      int col = h * DH + dt * 16 + (lane & 15);
      Og[((size_t)b * S + srow) * D + col] = (__bf16)(o[dt][r] * inv);
    }
  }
}

// ---------------------------------------------------------------------------
// LayerNorm(z + res) over D=1024; one block per row.
// ---------------------------------------------------------------------------
template <bool WRITE_BF16>
__global__ __launch_bounds__(256) void ln_row(
    const float* __restrict__ zin, const float* __restrict__ res,
    const float* __restrict__ gam, const float* __restrict__ bet,
    float* __restrict__ outf, __bf16* __restrict__ outb) {
  constexpr int D = 1024;
  const int row = blockIdx.x, tid = threadIdx.x;
  const float* zp = zin + (size_t)row * D;
  const float* rp = res + (size_t)row * D;

  float v[4], s = 0.f, ss = 0.f;
#pragma unroll
  for (int i = 0; i < 4; ++i) {
    int c = tid + 256 * i;
    float t = zp[c] + rp[c];
    v[i] = t; s += t; ss += t * t;
  }
  __shared__ float rs[256], rss[256];
  rs[tid] = s; rss[tid] = ss;
  __syncthreads();
  for (int o = 128; o > 0; o >>= 1) {
    if (tid < o) { rs[tid] += rs[tid + o]; rss[tid] += rss[tid + o]; }
    __syncthreads();
  }
  const float mu   = rs[0] * (1.0f / D);
  const float var  = rss[0] * (1.0f / D) - mu * mu;
  const float rstd = rsqrtf(var + 1e-5f);
#pragma unroll
  for (int i = 0; i < 4; ++i) {
    int c = tid + 256 * i;
    float o_ = (v[i] - mu) * rstd * gam[c] + bet[c];
    outf[(size_t)row * D + c] = o_;
    if (WRITE_BF16) outb[(size_t)row * D + c] = (__bf16)o_;
  }
}

// ---------------------------------------------------------------------------
// Host orchestration.
// ---------------------------------------------------------------------------
extern "C" void kernel_launch(void* const* d_in, const int* in_sizes, int n_in,
                              void* d_out, int out_size, void* d_ws, size_t ws_size,
                              hipStream_t stream) {
  (void)in_sizes; (void)n_in; (void)out_size; (void)ws_size;
  constexpr int B = 4, S = 2048, D = 1024, H = 16, DH = 64, DFF = 4096;
  const size_t BS = (size_t)B * S;  // 8192

  const float* x   = (const float*)d_in[0];
  const int*   kpm = (const int*)d_in[1];
  const float* Wq  = (const float*)d_in[2];
  const float* bq  = (const float*)d_in[3];
  const float* Wk  = (const float*)d_in[4];
  const float* bk  = (const float*)d_in[5];
  const float* Wv  = (const float*)d_in[6];
  const float* bv  = (const float*)d_in[7];
  const float* Wo  = (const float*)d_in[8];
  const float* bo  = (const float*)d_in[9];
  const float* lng = (const float*)d_in[10];
  const float* lnb = (const float*)d_in[11];
  const float* W1  = (const float*)d_in[12];
  const float* b1  = (const float*)d_in[13];
  const float* W2  = (const float*)d_in[14];
  const float* b2  = (const float*)d_in[15];

  char*  ws  = (char*)d_ws;
  size_t off = 0;
  auto carve = [&](size_t bytes) -> void* {
    void* p = ws + off;
    off += (bytes + 255) & ~(size_t)255;
    return p;
  };
  __bf16* xb   = (__bf16*)carve(BS * D * 2);
  __bf16* wqt  = (__bf16*)carve((size_t)H * DH * DH * 2);   // per-head [e][d]
  __bf16* wkt  = (__bf16*)carve((size_t)H * DH * DH * 2);
  __bf16* wvt  = (__bf16*)carve((size_t)H * DH * DH * 2);
  __bf16* wot  = (__bf16*)carve((size_t)D * D * 2);         // [n][k]
  __bf16* w1t  = (__bf16*)carve((size_t)DFF * D * 2);       // [DFF][D]
  __bf16* w2t  = (__bf16*)carve((size_t)D * DFF * 2);       // [D][DFF]
  __bf16* Qb   = (__bf16*)carve(BS * D * 2);                // (b,h,s,e)
  __bf16* Kb   = (__bf16*)carve(BS * D * 2);                // (b,h,s,e)
  __bf16* Vtb  = (__bf16*)carve(BS * D * 2);                // (b,h,e,s)  transposed
  __bf16* atb  = (__bf16*)carve(BS * D * 2);
  float*  yf   = (float*)carve(BS * D * 4);                 // reused for FFN2 out
  float*  h1f  = (float*)carve(BS * D * 4);
  __bf16* h1b  = (__bf16*)carve(BS * D * 2);
  __bf16* ffb  = (__bf16*)carve(BS * DFF * 2);

  // Conversions.
  {
    int grid = (int)((BS * D + 2047) / 2048);
    cvt_f32_bf16<<<grid, 256, 0, stream>>>(x, xb, BS * D);
  }
  cvt_transpose<<<dim3(DH / 32, DH / 32, H), 256, 0, stream>>>(Wq, wqt, DH, DH);
  cvt_transpose<<<dim3(DH / 32, DH / 32, H), 256, 0, stream>>>(Wk, wkt, DH, DH);
  cvt_transpose<<<dim3(DH / 32, DH / 32, H), 256, 0, stream>>>(Wv, wvt, DH, DH);
  cvt_transpose<<<dim3(D / 32, D / 32, 1), 256, 0, stream>>>(Wo, wot, D, D);
  cvt_transpose<<<dim3(DFF / 32, D / 32, 1), 256, 0, stream>>>(W1, w1t, D, DFF);
  cvt_transpose<<<dim3(D / 32, DFF / 32, 1), 256, 0, stream>>>(W2, w2t, DFF, D);

  // Per-head QKV GEMMs: z = b*H + h.
  {
    const dim3 g(S / 64, 1, B * H), blk(128);
    const long as1 = (long)S * D, as2 = DH;
    const long bs2 = (long)DH * DH, bias2 = DH;
    gemm_qkv<false><<<g, blk, 0, stream>>>(xb, wqt, bq, Qb, DH, D, DH, DH, H,
                                           as1, as2, bs2, bias2,
                                           (long)H * S * DH, (long)S * DH);
    gemm_qkv<false><<<g, blk, 0, stream>>>(xb, wkt, bk, Kb, DH, D, DH, DH, H,
                                           as1, as2, bs2, bias2,
                                           (long)H * S * DH, (long)S * DH);
    gemm_qkv<true><<<g, blk, 0, stream>>>(xb, wvt, bv, Vtb, DH, D, DH, S, H,
                                          as1, as2, bs2, bias2,
                                          (long)H * DH * S, (long)DH * S);
  }

  // Flash attention (TDM-staged K / V^T).
  attn_fwd<<<dim3(S / 64, B * H), dim3(128), 0, stream>>>(Qb, Kb, Vtb, kpm, atb);

  // Output projection: y = attn * Wo + bo (fp32).
  gemm128<false, false><<<dim3(BS / 128, D / 128), dim3(256), 0, stream>>>(
      atb, wot, bo, yf, D, D, D, D);

  // h1 = LN(y + x).
  ln_row<true><<<(unsigned)BS, 256, 0, stream>>>(yf, x, lng, lnb, h1f, h1b);

  // FFN1: relu(h1 * W1 + b1) -> bf16.
  gemm128<true, true><<<dim3(BS / 128, DFF / 128), dim3(256), 0, stream>>>(
      h1b, w1t, b1, ffb, D, D, D, DFF);

  // FFN2: ff * W2 + b2 -> fp32 (reuse yf).
  gemm128<false, false><<<dim3(BS / 128, D / 128), dim3(256), 0, stream>>>(
      ffb, w2t, b2, yf, DFF, DFF, DFF, D);

  // out = LN(ffn2 + h1).
  ln_row<false><<<(unsigned)BS, 256, 0, stream>>>(yf, h1f, lng, lnb,
                                                  (float*)d_out, nullptr);
}
